// DeepseekDecoderLayer_16587163697459
// MI455X (gfx1250) — matrix operand order repaired and checked by reading, not compiled, not run on hardware
//
#include <hip/hip_runtime.h>
#include <hip/hip_bf16.h>

// ---------------- types & helpers ----------------
typedef __attribute__((ext_vector_type(16))) __bf16 v16bf;
typedef __attribute__((ext_vector_type(8)))  float  v8f;
typedef __attribute__((ext_vector_type(4)))  unsigned u32x4;
typedef __attribute__((ext_vector_type(8)))  unsigned u32x8;

static constexpr int S_LEN = 1024;
static constexpr int H_DIM = 2048;
static constexpr int NHEAD = 16;
static constexpr int HEADD = 128;
static constexpr int NEXP  = 8;
static constexpr int FI_D  = 1408;
static constexpr int SFI_D = 2816;

__device__ __forceinline__ v8f v8f_zero() {
  v8f z = {0.f, 0.f, 0.f, 0.f, 0.f, 0.f, 0.f, 0.f};
  return z;
}

__device__ __forceinline__ v8f wmma_bf16(v16bf a, v16bf b, v8f c) {
  // D = A(16x32 bf16) * B(32x16 bf16) + C(16x16 f32)
  return __builtin_amdgcn_wmma_f32_16x16x32_bf16(false, a, false, b,
                                                 (short)0, c, false, false);
}

__device__ __forceinline__ unsigned pack_bf16x2(float a, float b) {
  union { __bf16 h[2]; unsigned u; } r;
  r.h[0] = (__bf16)a;
  r.h[1] = (__bf16)b;
  return r.u;
}

// ---- CDNA5 async global->LDS copy (ASYNCcnt) ----
__device__ __forceinline__ void async_ld_b128(unsigned lds_off, const void* gaddr) {
  asm volatile("global_load_async_to_lds_b128 %0, %1, off"
               :: "v"(lds_off), "v"(gaddr) : "memory");
}
__device__ __forceinline__ void wait_asynccnt0() {
  asm volatile("s_wait_asynccnt 0x0" ::: "memory");
}

// ---- CDNA5 Tensor Data Mover: 2D bf16 tile (tileRows x 32) global -> LDS ----
// LDS destination uses padded rows: 64B of data then 16B pad (stride 80B == 40 bf16),
// expressed via D# pad_interval=3 (16 DWORDs) / pad_amount=3 (4 DWORDs).
__device__ __forceinline__ void tdm_load_tile_bf16(const __bf16* gsrc, unsigned lds_off,
                                                   int K /*row stride (elems)*/,
                                                   int rows /*tensor_dim1*/) {
  unsigned long long ga = (unsigned long long)gsrc;
  u32x4 g0;
  g0[0] = 1u;                                             // count=1 (user descriptor)
  g0[1] = lds_off;                                        // lds_addr (bytes)
  g0[2] = (unsigned)(ga & 0xffffffffu);                   // global_addr[31:0]
  g0[3] = (unsigned)((ga >> 32) & 0x1ffffffu) | (2u << 30); // addr[56:32] | type=2
  unsigned td0 = (unsigned)K;
  u32x8 g1;
  g1[0] = (1u << 16)      // data_size = 1 (2 bytes)
        | (1u << 20)      // pad_enable
        | (3u << 22)      // pad_interval: 16 DWORDs between pads
        | (3u << 25);     // pad_amount: 4 DWORDs
  g1[1] = (td0 & 0xffffu) << 16;                          // tensor_dim0[15:0]
  g1[2] = ((td0 >> 16) & 0xffffu) | (((unsigned)rows & 0xffffu) << 16); // dim0 hi | dim1 lo
  g1[3] = (((unsigned)rows >> 16) & 0xffffu) | (32u << 16); // dim1 hi | tile_dim0=32
  g1[4] = 128u;                                           // tile_dim1=128, tile_dim2=0
  g1[5] = (unsigned)K;                                    // tensor_dim0_stride lo
  g1[6] = 0u;                                             // stride0 hi | stride1 lo
  g1[7] = 0u;                                             // stride1 hi
  asm volatile("tensor_load_to_lds %0, %1" :: "s"(g0), "s"(g1) : "memory");
}
__device__ __forceinline__ void wait_tensorcnt0() {
  asm volatile("s_wait_tensorcnt 0x0" ::: "memory");
}

// ---------------- RMSNorm (f32 in -> bf16 out) ----------------
__global__ __launch_bounds__(256)
void rmsnorm_kernel(const float* __restrict__ x, const float* __restrict__ w,
                    __bf16* __restrict__ out) {
  int t = blockIdx.x;
  __shared__ float red[8];
  const float* row = x + (size_t)t * H_DIM;
  float ss = 0.f;
  for (int i = threadIdx.x; i < H_DIM; i += 256) {
    float v = row[i];
    ss += v * v;
  }
  for (int o = 16; o > 0; o >>= 1) ss += __shfl_xor(ss, o, 32);
  if ((threadIdx.x & 31) == 0) red[threadIdx.x >> 5] = ss;
  __syncthreads();
  if (threadIdx.x == 0) {
    float s = 0.f;
    for (int i = 0; i < 8; i++) s += red[i];
    red[0] = rsqrtf(s / (float)H_DIM + 1e-6f);
  }
  __syncthreads();
  float rinv = red[0];
  __bf16* orow = out + (size_t)t * H_DIM;
  for (int i = threadIdx.x; i < H_DIM; i += 256)
    orow[i] = (__bf16)(row[i] * rinv * w[i]);
}

// ---------------- WMMA GEMM: C[M,N] = A[M,K](bf16) * B[N,K](f32)^T ----------------
// mode 0: C = acc ; mode 1: C = R + acc ; mode 2: C += rowscale[m]*acc
static constexpr int GBM = 128, GBN = 128, GBK = 32;
static constexpr int LSTR = 40;  // bf16 units per LDS row (padded, 16B aligned)

__global__ __launch_bounds__(256)
void gemm_bf16_kernel(const __bf16* __restrict__ A, const float* __restrict__ Bw,
                      float* __restrict__ C, const float* __restrict__ R,
                      const float* __restrict__ rowscale,
                      int M, int N, int K, int mode, int rsStride) {
  __shared__ __bf16 Alds[GBM * LSTR];
  __shared__ __bf16 Blds[GBN * LSTR];

  const int tid = threadIdx.x;
  const int nBlock = blockIdx.x * GBN;
  const int mBlock = blockIdx.y * GBM;
  const int lane = tid & 31;
  const int w = tid >> 5;
  const int wmOfs = (w & 3) * 32;    // 4 waves along M
  const int wnOfs = (w >> 2) * 64;   // 2 waves along N
  const int hi = lane >> 4;
  const int lm = lane & 15;

  v8f acc[2][4];
#pragma unroll
  for (int mi = 0; mi < 2; mi++)
#pragma unroll
    for (int ni = 0; ni < 4; ni++) acc[mi][ni] = v8f_zero();

  const int row2 = tid >> 1;  // 0..127 staging row
  const int half = tid & 1;   // 16-element half of the K slice
  const unsigned aldsOff = (unsigned)(size_t)(&Alds[0]);

  for (int k0 = 0; k0 < K; k0 += GBK) {
    // ---- stage A tile (bf16, 128x32) via Tensor Data Mover, wave 0 drives the DMA ----
    if (w == 0) {
      tdm_load_tile_bf16(A + (size_t)mBlock * K + k0, aldsOff, K, M);
      wait_tensorcnt0();
    }
    // ---- stage B tile (f32 -> bf16, 128x32), all 256 threads ----
    {
      const float* src = Bw + (size_t)(nBlock + row2) * K + k0 + half * 16;
      float4 f0 = ((const float4*)src)[0];
      float4 f1 = ((const float4*)src)[1];
      float4 f2 = ((const float4*)src)[2];
      float4 f3 = ((const float4*)src)[3];
      uint4 p0, p1;
      p0.x = pack_bf16x2(f0.x, f0.y); p0.y = pack_bf16x2(f0.z, f0.w);
      p0.z = pack_bf16x2(f1.x, f1.y); p0.w = pack_bf16x2(f1.z, f1.w);
      p1.x = pack_bf16x2(f2.x, f2.y); p1.y = pack_bf16x2(f2.z, f2.w);
      p1.z = pack_bf16x2(f3.x, f3.y); p1.w = pack_bf16x2(f3.z, f3.w);
      uint4* dst = (uint4*)&Blds[row2 * LSTR + half * 16];
      dst[0] = p0;
      dst[1] = p1;
      if (k0 + GBK < K)
        __builtin_prefetch(Bw + (size_t)(nBlock + row2) * K + k0 + GBK, 0, 3);
    }
    __syncthreads();

    // ---- fragments + WMMA ----
    v16bf afr[2], bfr[4];
#pragma unroll
    for (int mi = 0; mi < 2; mi++) {
      union { v16bf v; uint4 q[2]; } u;
      const __bf16* rb = &Alds[(wmOfs + mi * 16 + lm) * LSTR];
      u.q[0] = *(const uint4*)(rb + 8 * hi);        // K = 8*hi .. +7
      u.q[1] = *(const uint4*)(rb + 16 + 8 * hi);   // K = 16+8*hi .. +7
      afr[mi] = u.v;
    }
#pragma unroll
    for (int ni = 0; ni < 4; ni++) {
      union { v16bf v; uint4 q[2]; } u;
      const __bf16* rb = &Blds[(wnOfs + ni * 16 + lm) * LSTR];
      u.q[0] = *(const uint4*)(rb + 8 * hi);        // K = 8*hi .. +7
      u.q[1] = *(const uint4*)(rb + 16 + 8 * hi);   // K = 16+8*hi .. +7
      bfr[ni] = u.v;
    }
#pragma unroll
    for (int mi = 0; mi < 2; mi++)
#pragma unroll
      for (int ni = 0; ni < 4; ni++)
        acc[mi][ni] = wmma_bf16(afr[mi], bfr[ni], acc[mi][ni]);
    __syncthreads();
  }

  // ---- epilogue ----
#pragma unroll
  for (int mi = 0; mi < 2; mi++) {
#pragma unroll
    for (int ni = 0; ni < 4; ni++) {
      int n = nBlock + wnOfs + ni * 16 + lm;
#pragma unroll
      for (int r = 0; r < 8; r++) {
        int m = mBlock + wmOfs + mi * 16 + r + 8 * hi;
        size_t idx = (size_t)m * N + n;
        float v = acc[mi][ni][r];
        if (mode == 0) {
          C[idx] = v;
        } else if (mode == 1) {
          C[idx] = R[idx] + v;
        } else {
          float sc = rowscale ? rowscale[(size_t)m * rsStride] : 1.0f;
          C[idx] += sc * v;
        }
      }
    }
  }
}

// ---------------- RoPE: [S,H] f32 -> [NH,S,HD] bf16 (Q scaled by 1/sqrt(HD)) ----------------
__global__ __launch_bounds__(128)
void rope_kernel(const float* __restrict__ qf, const float* __restrict__ kf,
                 const float* __restrict__ vf, __bf16* __restrict__ qb,
                 __bf16* __restrict__ kb, __bf16* __restrict__ vb) {
  int s = blockIdx.x;
  int h = blockIdx.y;
  int d = threadIdx.x;  // 0..127
  int i = d & 63;
  float inv_freq = __expf(-(2.0f * (float)i / (float)HEADD) * __logf(10000.0f));
  float ang = (float)s * inv_freq;
  float c = cosf(ang), sn = sinf(ang);
  size_t base = (size_t)s * H_DIM + (size_t)h * HEADD;
  float qv = qf[base + d];
  float kv = kf[base + d];
  float qr = (d < 64) ? -qf[base + d + 64] : qf[base + d - 64];
  float kr = (d < 64) ? -kf[base + d + 64] : kf[base + d - 64];
  const float qscale = 0.08838834764831845f;  // 1/sqrt(128)
  size_t ob = ((size_t)h * S_LEN + s) * HEADD + d;
  qb[ob] = (__bf16)((qv * c + qr * sn) * qscale);
  kb[ob] = (__bf16)(kv * c + kr * sn);
  vb[ob] = (__bf16)vf[base + d];
}

// ---------------- Flash attention: one wave per (head, 16-row q tile) ----------------
__global__ __launch_bounds__(32)
void flash_attn_kernel(const __bf16* __restrict__ Q, const __bf16* __restrict__ Kc,
                       const __bf16* __restrict__ V, __bf16* __restrict__ Ob) {
  const int head = blockIdx.x >> 6;
  const int qt = blockIdx.x & 63;
  const int qbase = qt * 16;
  const int lane = threadIdx.x;
  const int hi = lane >> 4;
  const int lm = lane & 15;

  __shared__ __bf16 Plds[16 * 40];    // P tile 16x32 (padded)
  __shared__ __bf16 Vlds[32 * 136];   // V tile 32x128 (padded)

  const __bf16* Qh = Q + ((size_t)head * S_LEN + qbase) * HEADD;
  const __bf16* Kh = Kc + (size_t)head * S_LEN * HEADD;
  const __bf16* Vh = V + (size_t)head * S_LEN * HEADD;

  // Q A-fragments over HD=128 (4 K-steps of 32)
  v16bf qfr[4];
#pragma unroll
  for (int s = 0; s < 4; s++) {
    union { v16bf v; uint4 q[2]; } u;
    const __bf16* qrow = Qh + (size_t)lm * HEADD + s * 32;
    u.q[0] = *(const uint4*)(qrow + 8 * hi);
    u.q[1] = *(const uint4*)(qrow + 16 + 8 * hi);
    qfr[s] = u.v;
  }

  v8f Oacc[8];
#pragma unroll
  for (int i = 0; i < 8; i++) Oacc[i] = v8f_zero();
  float mrun[8], lrun[8];
#pragma unroll
  for (int r = 0; r < 8; r++) { mrun[r] = -1e30f; lrun[r] = 0.f; }

  const unsigned vldsRow = (unsigned)(size_t)(&Vlds[0]) + (unsigned)lane * (136u * 2u);

  const int nkb = (qbase + 16 + 31) >> 5;  // causal: key blocks of 32
  for (int kb = 0; kb < nkb; kb++) {
    const int kbase = kb * 32;
    // stage V tile (32 rows x 128 bf16) into LDS via async copy (ASYNCcnt)
    {
      const __bf16* vsrc = Vh + (size_t)(kbase + lane) * HEADD;
#pragma unroll
      for (int i = 0; i < 16; i++)
        async_ld_b128(vldsRow + (unsigned)i * 16u, vsrc + i * 8);
    }
    // S = Q * K^T : two 16x16 tiles (key cols kbase..+15, +16..+31)
    v8f s0 = v8f_zero(), s1 = v8f_zero();
#pragma unroll
    for (int s = 0; s < 4; s++) {
      union { v16bf v; uint4 q[2]; } b0, b1;
      const __bf16* k0p = Kh + (size_t)(kbase + lm) * HEADD + s * 32 + 8 * hi;
      b0.q[0] = *(const uint4*)(k0p);
      b0.q[1] = *(const uint4*)(k0p + 16);
      const __bf16* k1p = Kh + (size_t)(kbase + 16 + lm) * HEADD + s * 32 + 8 * hi;
      b1.q[0] = *(const uint4*)(k1p);
      b1.q[1] = *(const uint4*)(k1p + 16);
      s0 = wmma_bf16(qfr[s], b0.v, s0);
      s1 = wmma_bf16(qfr[s], b1.v, s1);
    }
    // online softmax per owned row (rows r+8*hi live across a 16-lane half)
    float p0v[8], p1v[8];
#pragma unroll
    for (int r = 0; r < 8; r++) {
      int qrow = qbase + r + 8 * hi;
      float v0 = s0[r];
      if (kbase + lm > qrow) v0 = -1e30f;
      float v1 = s1[r];
      if (kbase + 16 + lm > qrow) v1 = -1e30f;
      float mx = fmaxf(v0, v1);
      for (int o = 1; o < 16; o <<= 1) mx = fmaxf(mx, __shfl_xor(mx, o, 32));
      float mnew = fmaxf(mrun[r], mx);
      float scale = __expf(mrun[r] - mnew);
      float e0 = __expf(v0 - mnew);
      float e1 = __expf(v1 - mnew);
      float sum = e0 + e1;
      for (int o = 1; o < 16; o <<= 1) sum += __shfl_xor(sum, o, 32);
      lrun[r] = lrun[r] * scale + sum;
      mrun[r] = mnew;
      p0v[r] = e0;
      p1v[r] = e1;
#pragma unroll
      for (int t = 0; t < 8; t++) Oacc[t][r] *= scale;
    }
    // P (C-layout) -> LDS -> A-fragment layout
#pragma unroll
    for (int r = 0; r < 8; r++) {
      Plds[(r + 8 * hi) * 40 + lm] = (__bf16)p0v[r];
      Plds[(r + 8 * hi) * 40 + 16 + lm] = (__bf16)p1v[r];
    }
    union { v16bf v; uint4 q[2]; } pu;
    pu.q[0] = *(const uint4*)&Plds[lm * 40 + 8 * hi];
    pu.q[1] = *(const uint4*)&Plds[lm * 40 + 16 + 8 * hi];
    // O += P * V (wait for the async V tile first)
    wait_asynccnt0();
#pragma unroll
    for (int nt = 0; nt < 8; nt++) {
      union { v16bf v; __bf16 e[16]; } vb;
#pragma unroll
      for (int j = 0; j < 16; j++)
        vb.e[j] = Vlds[((j & 7) + 8 * hi + 16 * (j >> 3)) * 136 + nt * 16 + lm];
      Oacc[nt] = wmma_bf16(pu.v, vb.v, Oacc[nt]);
    }
  }
  // epilogue: normalize and store into [S,H] bf16 (head-interleaved columns)
#pragma unroll
  for (int nt = 0; nt < 8; nt++) {
#pragma unroll
    for (int r = 0; r < 8; r++) {
      int qrow = qbase + r + 8 * hi;
      float o = Oacc[nt][r] / lrun[r];
      Ob[(size_t)qrow * H_DIM + head * HEADD + nt * 16 + lm] = (__bf16)o;
    }
  }
}

// ---------------- Router ----------------
__global__ __launch_bounds__(256)
void router_logits_kernel(const __bf16* __restrict__ h2, const float* __restrict__ gw,
                          float* __restrict__ logits) {
  int t = blockIdx.x;
  int w = threadIdx.x >> 5;     // expert id (8 waves)
  int lane = threadIdx.x & 31;
  const __bf16* row = h2 + (size_t)t * H_DIM;
  const float* g = gw + (size_t)w * H_DIM;
  float s = 0.f;
  for (int k = lane; k < H_DIM; k += 32) s += (float)row[k] * g[k];
  for (int o = 16; o > 0; o >>= 1) s += __shfl_xor(s, o, 32);
  if (lane == 0) logits[t * NEXP + w] = s;
}

__global__ __launch_bounds__(256)
void router_combine_kernel(const float* __restrict__ logits, float* __restrict__ combine) {
  int t = blockIdx.x * 256 + threadIdx.x;
  if (t >= S_LEN) return;
  float l[NEXP];
  float mx = -1e30f;
  for (int e = 0; e < NEXP; e++) { l[e] = logits[t * NEXP + e]; mx = fmaxf(mx, l[e]); }
  float den = 0.f;
  for (int e = 0; e < NEXP; e++) { l[e] = __expf(l[e] - mx); den += l[e]; }
  for (int e = 0; e < NEXP; e++) l[e] /= den;
  int i0 = 0;
  for (int e = 1; e < NEXP; e++) if (l[e] > l[i0]) i0 = e;
  int i1 = (i0 == 0) ? 1 : 0;
  for (int e = 0; e < NEXP; e++) if (e != i0 && l[e] > l[i1]) i1 = e;
  float s = l[i0] + l[i1] + 1e-20f;
  for (int e = 0; e < NEXP; e++) combine[t * NEXP + e] = 0.f;
  combine[t * NEXP + i0] = l[i0] / s;
  combine[t * NEXP + i1] = l[i1] / s;
}

// ---------------- elementwise ----------------
__global__ __launch_bounds__(256)
void silu_mul_kernel(const float* __restrict__ g, const float* __restrict__ u,
                     __bf16* __restrict__ act, int n) {
  int i = blockIdx.x * 256 + threadIdx.x;
  if (i >= n) return;
  float gv = g[i];
  float s = gv / (1.f + __expf(-gv));
  act[i] = (__bf16)(s * u[i]);
}

__global__ __launch_bounds__(256)
void zero_kernel(float* __restrict__ y, int n) {
  int i = blockIdx.x * 256 + threadIdx.x;
  if (i < n) y[i] = 0.f;
}

__global__ __launch_bounds__(256)
void add_kernel(const float* __restrict__ x, const float* __restrict__ y,
                float* __restrict__ out, int n) {
  int i = blockIdx.x * 256 + threadIdx.x;
  if (i < n) out[i] = x[i] + y[i];
}

// ---------------- launch ----------------
extern "C" void kernel_launch(void* const* d_in, const int* in_sizes, int n_in,
                              void* d_out, int out_size, void* d_ws, size_t ws_size,
                              hipStream_t stream) {
  const float* hidden = (const float*)d_in[0];
  const float* ln1 = (const float*)d_in[1];
  const float* ln2 = (const float*)d_in[2];
  const float* wq = (const float*)d_in[3];
  const float* wk = (const float*)d_in[4];
  const float* wv = (const float*)d_in[5];
  const float* wo = (const float*)d_in[6];
  const float* gate_w = (const float*)d_in[7];
  const float* eg = (const float*)d_in[8];
  const float* eu = (const float*)d_in[9];
  const float* ed = (const float*)d_in[10];
  const float* sg = (const float*)d_in[11];
  const float* su = (const float*)d_in[12];
  const float* sd = (const float*)d_in[13];
  float* out = (float*)d_out;

  char* ws = (char*)d_ws;
  size_t off = 0;
  auto alloc = [&](size_t bytes) -> char* {
    char* p = ws + off;
    off += (bytes + 255) & ~(size_t)255;
    return p;
  };
  const size_t SH = (size_t)S_LEN * H_DIM;
  __bf16* h1b  = (__bf16*)alloc(SH * 2);
  float*  qf   = (float*)alloc(SH * 4);
  float*  kf   = (float*)alloc(SH * 4);
  float*  vf   = (float*)alloc(SH * 4);
  __bf16* qb   = (__bf16*)alloc(SH * 2);
  __bf16* kb   = (__bf16*)alloc(SH * 2);
  __bf16* vb   = (__bf16*)alloc(SH * 2);
  __bf16* attnb = (__bf16*)alloc(SH * 2);
  float*  xbuf = (float*)alloc(SH * 4);
  __bf16* h2b  = (__bf16*)alloc(SH * 2);
  float*  logits  = (float*)alloc((size_t)S_LEN * NEXP * 4);
  float*  combine = (float*)alloc((size_t)S_LEN * NEXP * 4);
  float*  ybuf = (float*)alloc(SH * 4);
  // aliases over dead buffers (launches serialize on the stream):
  // qf/kf/vf (24MB) are dead after RoPE -> g/u FFN intermediates (2 x 11.5MB)
  float*  gbuf = qf;
  float*  ubuf = (float*)((char*)qf + (size_t)S_LEN * SFI_D * 4);
  // qb/kb/vb (12MB) are dead after flash attention -> act buffer (<=5.75MB)
  __bf16* actb = qb;

  const dim3 gH(H_DIM / GBN, S_LEN / GBM);     // 16 x 8
  const dim3 gFI(FI_D / GBN, S_LEN / GBM);     // 11 x 8
  const dim3 gSFI(SFI_D / GBN, S_LEN / GBM);   // 22 x 8

  // 1. pre-attention RMSNorm
  rmsnorm_kernel<<<S_LEN, 256, 0, stream>>>(hidden, ln1, h1b);
  // 2. Q/K/V projections
  gemm_bf16_kernel<<<gH, 256, 0, stream>>>(h1b, wq, qf, nullptr, nullptr, S_LEN, H_DIM, H_DIM, 0, 0);
  gemm_bf16_kernel<<<gH, 256, 0, stream>>>(h1b, wk, kf, nullptr, nullptr, S_LEN, H_DIM, H_DIM, 0, 0);
  gemm_bf16_kernel<<<gH, 256, 0, stream>>>(h1b, wv, vf, nullptr, nullptr, S_LEN, H_DIM, H_DIM, 0, 0);
  // 3. RoPE + layout to [NH,S,HD] bf16
  rope_kernel<<<dim3(S_LEN, NHEAD), 128, 0, stream>>>(qf, kf, vf, qb, kb, vb);
  // 4. causal flash attention
  flash_attn_kernel<<<NHEAD * (S_LEN / 16), 32, 0, stream>>>(qb, kb, vb, attnb);
  // 5. output projection + residual
  gemm_bf16_kernel<<<gH, 256, 0, stream>>>(attnb, wo, xbuf, hidden, nullptr, S_LEN, H_DIM, H_DIM, 1, 0);
  // 6. post-attention RMSNorm
  rmsnorm_kernel<<<S_LEN, 256, 0, stream>>>(xbuf, ln2, h2b);
  // 7. router
  router_logits_kernel<<<S_LEN, 256, 0, stream>>>(h2b, gate_w, logits);
  router_combine_kernel<<<(S_LEN + 255) / 256, 256, 0, stream>>>(logits, combine);
  // 8. routed experts (dense, weighted by combine)
  zero_kernel<<<(int)((SH + 255) / 256), 256, 0, stream>>>(ybuf, (int)SH);
  for (int e = 0; e < NEXP; e++) {
    const float* egw = eg + (size_t)e * FI_D * H_DIM;
    const float* euw = eu + (size_t)e * FI_D * H_DIM;
    const float* edw = ed + (size_t)e * H_DIM * FI_D;
    gemm_bf16_kernel<<<gFI, 256, 0, stream>>>(h2b, egw, gbuf, nullptr, nullptr, S_LEN, FI_D, H_DIM, 0, 0);
    gemm_bf16_kernel<<<gFI, 256, 0, stream>>>(h2b, euw, ubuf, nullptr, nullptr, S_LEN, FI_D, H_DIM, 0, 0);
    silu_mul_kernel<<<(S_LEN * FI_D + 255) / 256, 256, 0, stream>>>(gbuf, ubuf, actb, S_LEN * FI_D);
    gemm_bf16_kernel<<<gH, 256, 0, stream>>>(actb, edw, ybuf, nullptr, combine + e, S_LEN, H_DIM, FI_D, 2, NEXP);
  }
  // 9. shared expert
  gemm_bf16_kernel<<<gSFI, 256, 0, stream>>>(h2b, sg, gbuf, nullptr, nullptr, S_LEN, SFI_D, H_DIM, 0, 0);
  gemm_bf16_kernel<<<gSFI, 256, 0, stream>>>(h2b, su, ubuf, nullptr, nullptr, S_LEN, SFI_D, H_DIM, 0, 0);
  silu_mul_kernel<<<(S_LEN * SFI_D + 255) / 256, 256, 0, stream>>>(gbuf, ubuf, actb, S_LEN * SFI_D);
  gemm_bf16_kernel<<<gH, 256, 0, stream>>>(actb, sd, ybuf, nullptr, nullptr, S_LEN, H_DIM, SFI_D, 2, 0);
  // 10. final residual
  add_kernel<<<(int)((SH + 255) / 256), 256, 0, stream>>>(xbuf, ybuf, out, (int)SH);
}
